// SocialAttention_82265803588251
// MI455X (gfx1250) — matrix operand order repaired and checked by reading, not compiled, not run on hardware
//
#include <hip/hip_runtime.h>
#include <hip/hip_bf16.h>

#define B_    32
#define A_    128
#define D_    256
#define H_    8
#define HD_   32
#define RELH_ 512

typedef __attribute__((ext_vector_type(16))) __bf16          v16bf;
typedef __attribute__((ext_vector_type(16))) unsigned short  v16us;
typedef __attribute__((ext_vector_type(8)))  unsigned short  v8us;
typedef __attribute__((ext_vector_type(8)))  float           v8f;
typedef __attribute__((ext_vector_type(2)))  float           v2f;

union BF16Frag { v16us u; v16bf b; };

// native f32 -> bf16 (RTNE) via hardware cvt
__device__ __forceinline__ unsigned short f2bf(float f) {
    union { __bf16 h; unsigned short u; } c;
    c.h = (__bf16)f;
    return c.u;
}

// ---- WMMA bf16 fragment loaders from bf16(u16) memory/LDS, 128-bit vector loads ----
// A (row-major): elems 0-7 = K[k0+8h..+7], elems 8-15 = K[k0+16+8h..+7]
__device__ __forceinline__ v16bf load_a_us(const unsigned short* row, int k0, int half) {
    v8us a0 = *(const v8us*)(row + k0 + half * 8);
    v8us a1 = *(const v8us*)(row + k0 + 16 + half * 8);
    BF16Frag t;
#pragma unroll
    for (int e = 0; e < 8; ++e) { t.u[e] = a0[e]; t.u[8 + e] = a1[e]; }
    return t.b;
}
// B = W^T (pass W rows): elems 0-7 = K[k0+8h..+7], elems 8-15 = K[k0+16+8h..+7]
__device__ __forceinline__ v16bf load_b_us(const unsigned short* row, int k0, int half) {
    v8us b0 = *(const v8us*)(row + k0 + half * 8);
    v8us b1 = *(const v8us*)(row + k0 + 16 + half * 8);
    BF16Frag t;
#pragma unroll
    for (int e = 0; e < 8; ++e) { t.u[e] = b0[e]; t.u[8 + e] = b1[e]; }
    return t.b;
}

// ==================== Kernel 0: f32 -> bf16 conversion pre-pass ====================
__global__ __launch_bounds__(256) void cvt_kernel(const float* __restrict__ in,
                                                  unsigned short* __restrict__ out, int n) {
    int i = blockIdx.x * 256 + threadIdx.x;
    if (i < n) out[i] = f2bf(in[i]);
}

// ========================== Kernel 1: QKV projection =============================
// qkv = x(4096x256) @ qkv_w^T(256->768) + b ; scatter bf16 into q/k/v as (B,H,A,HD)
// wave = 16 rows x 64 cols (4 N-tiles); K-loop double-buffered
__global__ __launch_bounds__(128) void qkv_kernel(
        const unsigned short* __restrict__ xbf, const unsigned short* __restrict__ wbf,
        const float* __restrict__ bias,
        unsigned short* __restrict__ q, unsigned short* __restrict__ k,
        unsigned short* __restrict__ v) {
    int lane = threadIdx.x & 31, wv = threadIdx.x >> 5;
    int r = lane & 15, half = lane >> 4;
    int n0 = blockIdx.x * 64;
    int m0 = (blockIdx.y * 4 + wv) * 16;
    v8f acc[4];
    const unsigned short* brow[4];
#pragma unroll
    for (int ct = 0; ct < 4; ++ct) {
        float bv = bias[n0 + ct * 16 + r];
#pragma unroll
        for (int i = 0; i < 8; ++i) acc[ct][i] = bv;
        brow[ct] = wbf + (size_t)(n0 + ct * 16 + r) * D_;
    }
    const unsigned short* arow = xbf + (size_t)(m0 + r) * D_;
    v16bf a0 = load_a_us(arow, 0, half);
    v16bf b0[4];
#pragma unroll
    for (int ct = 0; ct < 4; ++ct) b0[ct] = load_b_us(brow[ct], 0, half);
#pragma unroll
    for (int kt = 0; kt < 8; ++kt) {
        int kn = (kt < 7) ? (kt + 1) * 32 : 0;      // last iter: harmless reload of tile 0
        v16bf a1 = load_a_us(arow, kn, half);
        v16bf b1[4];
#pragma unroll
        for (int ct = 0; ct < 4; ++ct) b1[ct] = load_b_us(brow[ct], kn, half);
#pragma unroll
        for (int ct = 0; ct < 4; ++ct)
            acc[ct] = __builtin_amdgcn_wmma_f32_16x16x32_bf16(false, a0, false, b0[ct], (short)0, acc[ct], false, false);
        a0 = a1;
#pragma unroll
        for (int ct = 0; ct < 4; ++ct) b0[ct] = b1[ct];
    }
#pragma unroll
    for (int ct = 0; ct < 4; ++ct) {
        int n = n0 + ct * 16 + r;
        int sel = n >> 8;                 // 0=q 1=k 2=v
        int d = n & 255;
        int h = d >> 5, hd = d & 31;
        unsigned short* dst = (sel == 0) ? q : (sel == 1) ? k : v;
#pragma unroll
        for (int vi = 0; vi < 8; ++vi) {
            int m = m0 + vi + 8 * half;
            int bb = m >> 7, aa = m & 127;
            dst[(((size_t)(bb * H_ + h)) * A_ + aa) * HD_ + hd] = f2bf(acc[ct][vi]);
        }
    }
}

// ================== Kernel 2: fused relation MLP -> 16*sigmoid bias ==============
__global__ __launch_bounds__(32) void rel_kernel(
        const float* __restrict__ rel, const float* __restrict__ w1,
        const float* __restrict__ b1, const unsigned short* __restrict__ w2bf,
        float* __restrict__ biasbuf) {
    __shared__ unsigned short hid[16 * RELH_];   // 16x512 bf16 = 16KB
    int lane = threadIdx.x & 31;
    int r = lane & 15, half = lane >> 4;
    long M0 = (long)blockIdx.x * 16;
    v2f a2;                                      // 32-bit A 16x4: v0=K(2h), v1=K(2h+1)
    a2.x = rel[(M0 + r) * 4 + 2 * half];
    a2.y = rel[(M0 + r) * 4 + 2 * half + 1];
    for (int nt = 0; nt < 32; ++nt) {
        int n = nt * 16 + r;
        v2f b2;                                  // B = w1^T -> rows of w1
        b2.x = w1[n * 4 + 2 * half];
        b2.y = w1[n * 4 + 2 * half + 1];
        float bb = b1[n];
        v8f acc;
#pragma unroll
        for (int i = 0; i < 8; ++i) acc[i] = bb;
        acc = __builtin_amdgcn_wmma_f32_16x16x4_f32(false, a2, false, b2, (short)0, acc, false, false);
#pragma unroll
        for (int vi = 0; vi < 8; ++vi) {
            float hv = acc[vi] > 0.0f ? acc[vi] : 0.0f;   // relu
            hid[(vi + 8 * half) * RELH_ + n] = f2bf(hv);
        }
    }
    __syncthreads();
    BF16Frag zf;
#pragma unroll
    for (int e = 0; e < 16; ++e) zf.u[e] = 0;
    // two accumulators break the 16-deep WMMA->WMMA RAW chain; double-buffered frags
    v8f accE = {}, accO = {};
    v16bf a0 = load_a_us(hid + r * RELH_, 0, half);
    v16bf bb0 = (r < 8) ? load_b_us(w2bf + (size_t)r * RELH_, 0, half) : zf.b;
#pragma unroll
    for (int kt = 0; kt < 16; ++kt) {
        int kn = (kt < 15) ? (kt + 1) * 32 : 0;
        v16bf a1 = load_a_us(hid + r * RELH_, kn, half);
        v16bf bb1 = (r < 8) ? load_b_us(w2bf + (size_t)r * RELH_, kn, half) : zf.b;
        if (kt & 1)
            accO = __builtin_amdgcn_wmma_f32_16x16x32_bf16(false, a0, false, bb0, (short)0, accO, false, false);
        else
            accE = __builtin_amdgcn_wmma_f32_16x16x32_bf16(false, a0, false, bb0, (short)0, accE, false, false);
        a0 = a1; bb0 = bb1;
    }
    int b = (int)(M0 >> 14);
    int rem = (int)(M0 & 16383);
    int i = rem >> 7, j0 = rem & 127;
    if (r < 8) {                                  // col r = head h
#pragma unroll
        for (int vi = 0; vi < 8; ++vi) {
            int j = j0 + vi + 8 * half;
            float e = accE[vi] + accO[vi];
            float s = 16.0f / (1.0f + __expf(-e));
            biasbuf[(((size_t)(b * H_ + r)) * A_ + i) * A_ + j] = s;
        }
    }
}

// ======= Kernel 3: fused attention per (b,h): QK^T + bias + mask, softmax, PV =======
__global__ __launch_bounds__(256) void attn_kernel(
        const unsigned short* __restrict__ q, const unsigned short* __restrict__ kk,
        const unsigned short* __restrict__ vv, const float* __restrict__ biasbuf,
        const unsigned char* __restrict__ mask, float* __restrict__ out) {
    __shared__ unsigned short kst[A_ * HD_];      // K tile bf16 (row-major)  8KB, async-staged
    __shared__ unsigned short vt[HD_ * A_];       // V^T bf16 [d][j]          8KB
    __shared__ unsigned short ps[8 * 16 * A_];    // P strips bf16, per-wave 16x128  32KB
    int bh = blockIdx.x; int b = bh >> 3, h = bh & 7;
    const size_t base = (size_t)bh * (A_ * HD_);
    const unsigned short* qb = q  + base;
    const unsigned short* kb = kk + base;
    const unsigned short* vb = vv + base;
    const float* bb = biasbuf + (size_t)bh * (A_ * A_);
    int t = threadIdx.x;
    // --- CDNA5 async memory->LDS staging of K (no VGPR round-trip, ASYNCcnt) ---
    {
        unsigned lds0 = (unsigned)(size_t)(&kst[0]);          // low 32 bits = LDS offset
        for (int dw = t; dw < (A_ * HD_) / 2; dw += 256) {    // 2048 dwords
            unsigned dst = lds0 + dw * 4;
            const unsigned short* src = kb + dw * 2;
            asm volatile("global_load_async_to_lds_b32 %0, %1, off"
                         :: "v"(dst), "v"(src) : "memory");
        }
    }
    // --- V^T staged through VGPRs (transpose) ---
    for (int idx = t; idx < A_ * HD_; idx += 256) {
        int j = idx >> 5, d = idx & 31;
        vt[d * A_ + j] = vb[idx];
    }
    asm volatile("s_wait_asynccnt 0x0" ::: "memory");
    __syncthreads();
    int lane = t & 31, wv = t >> 5;
    int r = lane & 15, half = lane >> 4;
    int i0 = wv * 16;                              // wave owns query rows i0..i0+15
    const float scale = 0.17677669529663687f;      // 1/sqrt(HD)
    v16bf aq = load_a_us(qb + (size_t)(i0 + r) * HD_, 0, half);  // invariant across nt
    v8f acc[8];
#pragma unroll
    for (int nt = 0; nt < 8; ++nt) {
        int n0 = nt * 16;
        v16bf bf = load_b_us(kst + (size_t)(n0 + r) * HD_, 0, half);
        v8f c = {};
        c = __builtin_amdgcn_wmma_f32_16x16x32_bf16(false, aq, false, bf, (short)0, c, false, false);
        int j = n0 + r;
        bool mskd = mask[b * A_ + j] != 0;
#pragma unroll
        for (int vi = 0; vi < 8; ++vi) {
            int i = i0 + vi + 8 * half;
            c[vi] = mskd ? -__builtin_inff() : fmaf(c[vi], scale, bb[(size_t)i * A_ + j]);
        }
        acc[nt] = c;
    }
    // softmax: row m=vi+8*half lives in a 16-lane half-group -> shfl_xor(8,4,2,1)
#pragma unroll
    for (int vi = 0; vi < 8; ++vi) {
        float mx = acc[0][vi];
#pragma unroll
        for (int nt = 1; nt < 8; ++nt) mx = fmaxf(mx, acc[nt][vi]);
#pragma unroll
        for (int off = 8; off >= 1; off >>= 1) mx = fmaxf(mx, __shfl_xor(mx, off, 32));
        float sum = 0.0f;
#pragma unroll
        for (int nt = 0; nt < 8; ++nt) { float e = __expf(acc[nt][vi] - mx); acc[nt][vi] = e; sum += e; }
#pragma unroll
        for (int off = 8; off >= 1; off >>= 1) sum += __shfl_xor(sum, off, 32);
        float is = 1.0f / sum;
#pragma unroll
        for (int nt = 0; nt < 8; ++nt) acc[nt][vi] *= is;
    }
    unsigned short* strip = ps + wv * 16 * A_;     // C-layout -> A-layout transpose via LDS
#pragma unroll
    for (int nt = 0; nt < 8; ++nt) {
        int j = nt * 16 + r;
#pragma unroll
        for (int vi = 0; vi < 8; ++vi)
            strip[(vi + 8 * half) * A_ + j] = f2bf(acc[nt][vi]);
    }
    __syncthreads();
    v8f oc[2] = {{}, {}};
#pragma unroll
    for (int kt = 0; kt < 4; ++kt) {               // K-outer: A-frag reused for both d-tiles
        int k0 = kt * 32;
        v16bf a = load_a_us(strip + r * A_, k0, half);
#pragma unroll
        for (int dt = 0; dt < 2; ++dt) {
            v16bf bf = load_b_us(vt + (dt * 16 + r) * A_, k0, half);
            oc[dt] = __builtin_amdgcn_wmma_f32_16x16x32_bf16(false, a, false, bf, (short)0, oc[dt], false, false);
        }
    }
#pragma unroll
    for (int dt = 0; dt < 2; ++dt) {
        int d = dt * 16 + r;
#pragma unroll
        for (int vi = 0; vi < 8; ++vi) {
            int i = i0 + vi + 8 * half;
            out[((size_t)(b * A_ + i)) * D_ + h * HD_ + d] = oc[dt][vi];
        }
    }
}

// ======= Kernel 4a: x1 = LayerNorm(attn_out + x) ; writes f32 + bf16 copies =======
__global__ __launch_bounds__(256) void ln1_kernel(
        const float* __restrict__ ao, const float* __restrict__ x,
        const float* __restrict__ g, const float* __restrict__ be,
        float* __restrict__ x1, unsigned short* __restrict__ x1bf) {
    int row = blockIdx.x * 8 + (threadIdx.x >> 5);
    int lane = threadIdx.x & 31;
    const float* pa = ao + (size_t)row * D_;
    const float* px = x  + (size_t)row * D_;
    float vals[8]; float s = 0.0f, s2 = 0.0f;
#pragma unroll
    for (int i = 0; i < 8; ++i) {
        float v = pa[lane + 32 * i] + px[lane + 32 * i];
        vals[i] = v; s += v; s2 += v * v;
    }
#pragma unroll
    for (int off = 16; off >= 1; off >>= 1) { s += __shfl_xor(s, off, 32); s2 += __shfl_xor(s2, off, 32); }
    float mu = s * (1.0f / D_);
    float rs = rsqrtf(s2 * (1.0f / D_) - mu * mu + 1e-5f);
    float* po = x1 + (size_t)row * D_;
    unsigned short* pb = x1bf + (size_t)row * D_;
#pragma unroll
    for (int i = 0; i < 8; ++i) {
        int c = lane + 32 * i;
        float o = (vals[i] - mu) * rs * g[c] + be[c];
        po[c] = o;
        pb[c] = f2bf(o);
    }
}

// ===== Kernel 4b: ffh = relu(x1 @ ff_w1^T + b1) bf16 (4096x768); 4 N-tiles/wave =====
__global__ __launch_bounds__(128) void ff1_kernel(
        const unsigned short* __restrict__ x1bf, const unsigned short* __restrict__ wbf,
        const float* __restrict__ bias, unsigned short* __restrict__ hbuf) {
    int lane = threadIdx.x & 31, wv = threadIdx.x >> 5;
    int r = lane & 15, half = lane >> 4;
    int n0 = blockIdx.x * 64;
    int m0 = (blockIdx.y * 4 + wv) * 16;
    v8f acc[4];
    const unsigned short* brow[4];
#pragma unroll
    for (int ct = 0; ct < 4; ++ct) {
        float bv = bias[n0 + ct * 16 + r];
#pragma unroll
        for (int i = 0; i < 8; ++i) acc[ct][i] = bv;
        brow[ct] = wbf + (size_t)(n0 + ct * 16 + r) * D_;
    }
    const unsigned short* arow = x1bf + (size_t)(m0 + r) * D_;
    v16bf a0 = load_a_us(arow, 0, half);
    v16bf b0[4];
#pragma unroll
    for (int ct = 0; ct < 4; ++ct) b0[ct] = load_b_us(brow[ct], 0, half);
#pragma unroll
    for (int kt = 0; kt < 8; ++kt) {
        int kn = (kt < 7) ? (kt + 1) * 32 : 0;
        v16bf a1 = load_a_us(arow, kn, half);
        v16bf b1[4];
#pragma unroll
        for (int ct = 0; ct < 4; ++ct) b1[ct] = load_b_us(brow[ct], kn, half);
#pragma unroll
        for (int ct = 0; ct < 4; ++ct)
            acc[ct] = __builtin_amdgcn_wmma_f32_16x16x32_bf16(false, a0, false, b0[ct], (short)0, acc[ct], false, false);
        a0 = a1;
#pragma unroll
        for (int ct = 0; ct < 4; ++ct) b0[ct] = b1[ct];
    }
#pragma unroll
    for (int ct = 0; ct < 4; ++ct) {
#pragma unroll
        for (int vi = 0; vi < 8; ++vi) {
            int m = m0 + vi + 8 * half;
            float hv = acc[ct][vi] > 0.0f ? acc[ct][vi] : 0.0f;
            hbuf[(size_t)m * 768 + n0 + ct * 16 + r] = f2bf(hv);
        }
    }
}

// ====== Kernel 4c: out = LayerNorm(ffh @ ff_w2^T + b2 + x1) — fused GEMM+LN2 ======
__global__ __launch_bounds__(128) void ff2_kernel(
        const unsigned short* __restrict__ hbuf, const unsigned short* __restrict__ w2bf,
        const float* __restrict__ b2, const float* __restrict__ x1,
        const float* __restrict__ g, const float* __restrict__ be,
        float* __restrict__ out) {
    __shared__ float y[16 * D_];                  // 16KB
    __shared__ float mu_s[16], rs_s[16];
    int lane = threadIdx.x & 31, wv = threadIdx.x >> 5;
    int r = lane & 15, half = lane >> 4;
    int m0 = blockIdx.x * 16;
    v8f acc[4];
    const unsigned short* brow[4];
#pragma unroll
    for (int ct = 0; ct < 4; ++ct) {
        float bv = b2[(wv * 4 + ct) * 16 + r];
#pragma unroll
        for (int i = 0; i < 8; ++i) acc[ct][i] = bv;
        brow[ct] = w2bf + (size_t)((wv * 4 + ct) * 16 + r) * 768;
    }
    const unsigned short* arow = hbuf + (size_t)(m0 + r) * 768;
    v16bf a0 = load_a_us(arow, 0, half);
    v16bf b0[4];
#pragma unroll
    for (int ct = 0; ct < 4; ++ct) b0[ct] = load_b_us(brow[ct], 0, half);
    for (int k0 = 0; k0 < 768; k0 += 32) {        // double-buffered K loop
        int kn = (k0 + 32 < 768) ? k0 + 32 : 0;
        v16bf a1 = load_a_us(arow, kn, half);
        v16bf b1[4];
#pragma unroll
        for (int ct = 0; ct < 4; ++ct) b1[ct] = load_b_us(brow[ct], kn, half);
#pragma unroll
        for (int ct = 0; ct < 4; ++ct)
            acc[ct] = __builtin_amdgcn_wmma_f32_16x16x32_bf16(false, a0, false, b0[ct], (short)0, acc[ct], false, false);
        a0 = a1;
#pragma unroll
        for (int ct = 0; ct < 4; ++ct) b0[ct] = b1[ct];
    }
#pragma unroll
    for (int ct = 0; ct < 4; ++ct) {
        int n0 = (wv * 4 + ct) * 16;
#pragma unroll
        for (int vi = 0; vi < 8; ++vi) {
            int mr = vi + 8 * half;
            y[mr * D_ + n0 + r] = acc[ct][vi] + x1[(size_t)(m0 + mr) * D_ + n0 + r];
        }
    }
    __syncthreads();
    int t = threadIdx.x;
    if (t < 16) {
        float s = 0.0f, s2 = 0.0f;
        const float* row = y + t * D_;
        for (int c = 0; c < D_; ++c) { float v = row[c]; s += v; s2 += v * v; }
        float mu = s * (1.0f / D_);
        mu_s[t] = mu;
        rs_s[t] = rsqrtf(s2 * (1.0f / D_) - mu * mu + 1e-5f);
    }
    __syncthreads();
    for (int idx = t; idx < 16 * D_; idx += 128) {
        int mr = idx >> 8, c = idx & 255;
        out[(size_t)(m0 + mr) * D_ + c] = (y[idx] - mu_s[mr]) * rs_s[mr] * g[c] + be[c];
    }
}

extern "C" void kernel_launch(void* const* d_in, const int* in_sizes, int n_in,
                              void* d_out, int out_size, void* d_ws, size_t ws_size,
                              hipStream_t stream) {
    const float* x      = (const float*)d_in[0];
    const unsigned char* mask = (const unsigned char*)d_in[1];
    const float* rel    = (const float*)d_in[2];
    const float* qkv_w  = (const float*)d_in[3];
    const float* qkv_b  = (const float*)d_in[4];
    const float* rel_w1 = (const float*)d_in[5];
    const float* rel_b1 = (const float*)d_in[6];
    const float* rel_w2 = (const float*)d_in[7];
    const float* ff_w1  = (const float*)d_in[8];
    const float* ff_b1  = (const float*)d_in[9];
    const float* ff_w2  = (const float*)d_in[10];
    const float* ff_b2  = (const float*)d_in[11];
    const float* ln1_g  = (const float*)d_in[12];
    const float* ln1_b  = (const float*)d_in[13];
    const float* ln2_g  = (const float*)d_in[14];
    const float* ln2_b  = (const float*)d_in[15];
    float* out = (float*)d_out;

    char* ws = (char*)d_ws;
    const size_t MiB = 1024 * 1024;
    const size_t KiB = 1024;
    unsigned short* qbf    = (unsigned short*)(ws + 0 * MiB);   // 2 MiB (B,H,A,HD) bf16
    unsigned short* kbf    = (unsigned short*)(ws + 2 * MiB);   // 2 MiB
    unsigned short* vbf    = (unsigned short*)(ws + 4 * MiB);   // 2 MiB
    float*          relbuf = (float*)(ws + 6 * MiB);            // 16 MiB (B,H,A,A)
    float*          aobuf  = (float*)(ws + 22 * MiB);           // 4 MiB attn out f32
    float*          x1buf  = (float*)(ws + 26 * MiB);           // 4 MiB
    unsigned short* x1bf   = (unsigned short*)(ws + 30 * MiB);  // 2 MiB
    unsigned short* ffh    = (unsigned short*)(ws + 32 * MiB);  // 6 MiB (4096x768)
    unsigned short* xbf    = (unsigned short*)(ws + 38 * MiB);  // 2 MiB
    unsigned short* qkvwbf = (unsigned short*)(ws + 40 * MiB);              // 384 KiB
    unsigned short* ffw1bf = (unsigned short*)(ws + 40 * MiB + 512 * KiB);  // 384 KiB
    unsigned short* ffw2bf = (unsigned short*)(ws + 41 * MiB);              // 384 KiB
    unsigned short* rw2bf  = (unsigned short*)(ws + 41 * MiB + 512 * KiB);  // 8 KiB

    const int n_x = 4096 * 256, n_w = 768 * 256, n_rw2 = 8 * 512;
    cvt_kernel<<<dim3((n_x + 255) / 256), 256, 0, stream>>>(x, xbf, n_x);
    cvt_kernel<<<dim3((n_w + 255) / 256), 256, 0, stream>>>(qkv_w, qkvwbf, n_w);
    cvt_kernel<<<dim3((n_w + 255) / 256), 256, 0, stream>>>(ff_w1, ffw1bf, n_w);
    cvt_kernel<<<dim3((n_w + 255) / 256), 256, 0, stream>>>(ff_w2, ffw2bf, n_w);
    cvt_kernel<<<dim3((n_rw2 + 255) / 256), 256, 0, stream>>>(rel_w2, rw2bf, n_rw2);

    qkv_kernel<<<dim3(12, 64), 128, 0, stream>>>(xbf, qkvwbf, qkv_b, qbf, kbf, vbf);
    rel_kernel<<<dim3(32768), 32, 0, stream>>>(rel, rel_w1, rel_b1, rw2bf, relbuf);
    attn_kernel<<<dim3(256), 256, 0, stream>>>(qbf, kbf, vbf, relbuf, mask, aobuf);
    ln1_kernel<<<dim3(512), 256, 0, stream>>>(aobuf, x, ln1_g, ln1_b, x1buf, x1bf);
    ff1_kernel<<<dim3(12, 64), 128, 0, stream>>>(x1bf, ffw1bf, ff_b1, ffh);
    ff2_kernel<<<dim3(256), 128, 0, stream>>>(ffh, ffw2bf, ff_b2, x1buf, ln2_g, ln2_b, out);
}
